// GPT2Attention_26439818674263
// MI455X (gfx1250) — matrix operand order repaired and checked by reading, not compiled, not run on hardware
//
#include <hip/hip_runtime.h>

// ---------------------------------------------------------------------------
// GPT-2 attention forward for MI455X (gfx1250, wave32, WMMA bf16 16x16x32).
// Pipeline: cvt(X,W) -> QKV GEMM (TDM double-buffered, bf16 in, f32 acc)
//           -> flash attention (causal, online softmax, TDM-staged Q/K)
//           -> output projection GEMM (f32 out, bias fused).
// ---------------------------------------------------------------------------

typedef __bf16 bf16x16 __attribute__((ext_vector_type(16)));
typedef __bf16 bf16x8  __attribute__((ext_vector_type(8)));
typedef float  f32x8   __attribute__((ext_vector_type(8)));
typedef unsigned int u32;
typedef u32  u32x4 __attribute__((ext_vector_type(4)));
typedef int  i32x8 __attribute__((ext_vector_type(8)));
typedef int  i32x4 __attribute__((ext_vector_type(4)));

#if __has_builtin(__builtin_amdgcn_tensor_load_to_lds) && \
    __has_builtin(__builtin_amdgcn_s_wait_tensorcnt)
#define USE_TDM 1
#else
#define USE_TDM 0
#endif

union Frag {
    bf16x16 v;
    bf16x8  h[2];
};

__device__ __forceinline__ f32x8 wmma_bf16(bf16x16 a, bf16x16 b, f32x8 c) {
    // D = A(16x32 bf16) * B(32x16 bf16) + C(16x16 f32)
    return __builtin_amdgcn_wmma_f32_16x16x32_bf16(
        /*neg_a=*/false, a, /*neg_b=*/false, b,
        /*c_mod=*/(short)0, c, /*reuse_a=*/false, /*reuse_b=*/false);
}

__device__ __forceinline__ unsigned short f2bf(float f) {
    unsigned int u = __float_as_uint(f);
    unsigned int r = u + 0x7FFFu + ((u >> 16) & 1u); // round-to-nearest-even
    return (unsigned short)(r >> 16);
}

__device__ __forceinline__ u32 lds_byte_off(const void* p) {
    // Generic LDS address: low 32 bits are the LDS byte offset (aperture map).
    return (u32)(unsigned long long)p;
}

#if USE_TDM
// Issue a 2D TDM tile load (bf16 elements) with LDS row padding.
// Tile: tile_d1 rows x tile_d0 elements; global row stride = stride0 elems.
// pad_iv: interval code (2^(c+1) dwords), pad_am: amount code (c+1 dwords).
__device__ __forceinline__ void tdm_load_2d(const void* gptr, u32 lds_off,
                                            u32 tens_d0, u32 tens_d1,
                                            u32 tile_d0, u32 tile_d1,
                                            u32 stride0, u32 pad_iv, u32 pad_am) {
    unsigned long long ga = (unsigned long long)gptr;
    u32x4 g0;
    g0[0] = 1u;                                  // count=1 (valid user descriptor)
    g0[1] = lds_off;                             // lds_addr (bytes)
    g0[2] = (u32)(ga & 0xFFFFFFFFu);             // global_addr[31:0]
    g0[3] = (u32)((ga >> 32) & 0x1FFFFFFu) | (2u << 30); // addr[56:32] | type=2
    i32x8 g1;
    g1[0] = (int)((1u << 16) |                   // data_size = 1 -> 2 bytes
                  (1u << 20) |                   // pad_enable
                  (pad_iv << 22) | (pad_am << 25));
    g1[1] = (int)((tens_d0 & 0xFFFFu) << 16);    // abar=0 | tensor_dim0[15:0]
    g1[2] = (int)((tens_d0 >> 16) | ((tens_d1 & 0xFFFFu) << 16));
    g1[3] = (int)((tens_d1 >> 16) | (tile_d0 << 16));
    g1[4] = (int)(tile_d1 & 0xFFFFu);            // tile_dim1 | tile_dim2=0
    g1[5] = (int)stride0;                        // tensor_dim0_stride[31:0]
    g1[6] = 0;                                   // stride0[47:32] | dim1_stride lo
    g1[7] = 0;
    i32x4 z4 = {0, 0, 0, 0};                     // groups 2/3 unused (2D tile)
    i32x8 z8 = {0, 0, 0, 0, 0, 0, 0, 0};         // extra group (6-arg toolchain)
    __builtin_amdgcn_tensor_load_to_lds(g0, g1, z4, z4, z8, 0);
}
#endif

// ---------------------------------------------------------------------------
// Conversion kernels
// ---------------------------------------------------------------------------
__global__ __launch_bounds__(256) void cvt_f32_to_bf16(
    const float* __restrict__ in, unsigned short* __restrict__ out, int n) {
    int i = blockIdx.x * 256 + threadIdx.x;
    if (i < n) out[i] = f2bf(in[i]);
}

// out[n*1024 + k] = bf16(in[k*N + n]); K fixed at 1024 for both weights.
__global__ __launch_bounds__(256) void cvt_transpose_1024(
    const float* __restrict__ in, unsigned short* __restrict__ out, int N) {
    int i = blockIdx.x * 256 + threadIdx.x; // i = n*1024 + k
    if (i < N * 1024) {
        int k = i & 1023;
        int n = i >> 10;
        out[i] = f2bf(in[(size_t)k * N + n]);
    }
}

// ---------------------------------------------------------------------------
// Shared WMMA micro-kernel for one 32-K step on a 128x128 tile.
// Wave grid 4(M) x 2(N): wave tile 32x64 -> 8 accumulators, 8 wmma / step.
// ---------------------------------------------------------------------------
#define GEMM_LDT 40  // padded row stride in shorts (80B: 16B-aligned, 20 banks)

__device__ __forceinline__ void gemm_step(const unsigned short* __restrict__ sA,
                                          const unsigned short* __restrict__ sB,
                                          f32x8 (&acc)[2][4], int waveM, int waveN,
                                          int lo, int half) {
    // A fragments: ISA 16-bit A layout -> chunks at K = half*8, 16+half*8
    Frag af[2];
#pragma unroll
    for (int mf = 0; mf < 2; ++mf) {
        int r = waveM * 32 + mf * 16 + lo;
        af[mf].h[0] = *reinterpret_cast<const bf16x8*>(&sA[r * GEMM_LDT + half * 8]);
        af[mf].h[1] = *reinterpret_cast<const bf16x8*>(&sA[r * GEMM_LDT + 16 + half * 8]);
    }
    // B fragments: lane = column n, chunks at K = half*8, 16+half*8
    Frag bfr[4];
#pragma unroll
    for (int nf = 0; nf < 4; ++nf) {
        int r = waveN * 64 + nf * 16 + lo;
        bfr[nf].h[0] = *reinterpret_cast<const bf16x8*>(&sB[r * GEMM_LDT + half * 8]);
        bfr[nf].h[1] = *reinterpret_cast<const bf16x8*>(&sB[r * GEMM_LDT + 16 + half * 8]);
    }
#pragma unroll
    for (int mf = 0; mf < 2; ++mf)
#pragma unroll
        for (int nf = 0; nf < 4; ++nf)
            acc[mf][nf] = wmma_bf16(af[mf].v, bfr[nf].v, acc[mf][nf]);
}

// ---------------------------------------------------------------------------
// bf16 GEMM: C[M][N] = A[M][K] * Bt[N][K]^T + bias   (Bt pre-transposed)
// Block: 256 threads (8 waves), tile 128x128, K-step 32 (one WMMA step).
// TDM path: double-buffered LDS, tiles DMAed by the TDM with hardware row
// padding (16 dwords data + 4 dwords pad -> 40-short stride), overlapped
// with WMMA compute via TENSORcnt.
// ---------------------------------------------------------------------------
template <bool OUT_F32>
__global__ __launch_bounds__(256) void gemm_bf16_wmma(
    const unsigned short* __restrict__ A,   // bf16 [M][K]
    const unsigned short* __restrict__ Bt,  // bf16 [N][K]
    const float* __restrict__ bias,         // [N]
    float* __restrict__ outF,               // if OUT_F32
    unsigned short* __restrict__ outB,      // if !OUT_F32 (bf16)
    int M, int N, int K) {
    const int tid  = threadIdx.x;
    const int lane = tid & 31;
    const int lo   = lane & 15;
    const int half = lane >> 4;
    const int wave  = tid >> 5;
    const int waveM = wave >> 1; // 0..3
    const int waveN = wave & 1;  // 0..1
    const int m0 = blockIdx.y * 128;
    const int n0 = blockIdx.x * 128;

    f32x8 acc[2][4] = {};

#if USE_TDM
    __shared__ __align__(16) unsigned short lA[2][128 * GEMM_LDT];
    __shared__ __align__(16) unsigned short lB[2][128 * GEMM_LDT];

    const unsigned short* Ab = &A[(size_t)m0 * K];
    const unsigned short* Bb = &Bt[(size_t)n0 * K];
    // Prologue: DMA tile 0. pad codes: interval 3 -> 16 dwords, amount 3 -> 4.
    if (tid < 32) {
        tdm_load_2d(Ab, lds_byte_off(lA[0]), 32, 128, 32, 128, (u32)K, 3, 3);
        tdm_load_2d(Bb, lds_byte_off(lB[0]), 32, 128, 32, 128, (u32)K, 3, 3);
    }
    int cur = 0;
    for (int kb = 0; kb < K; kb += 32) {
        if (tid < 32) {
            if (kb + 32 < K) {
                // Prefetch next tile into the other buffer, then wait for the
                // current tile only (TDM ops complete in order: <=2 pending).
                tdm_load_2d(Ab + kb + 32, lds_byte_off(lA[cur ^ 1]),
                            32, 128, 32, 128, (u32)K, 3, 3);
                tdm_load_2d(Bb + kb + 32, lds_byte_off(lB[cur ^ 1]),
                            32, 128, 32, 128, (u32)K, 3, 3);
                __builtin_amdgcn_s_wait_tensorcnt(2);
            } else {
                __builtin_amdgcn_s_wait_tensorcnt(0);
            }
        }
        __syncthreads();                 // tile `cur` visible to all waves
        gemm_step(lA[cur], lB[cur], acc, waveM, waveN, lo, half);
        __syncthreads();                 // all reads done before buffer reuse
        cur ^= 1;
    }
#else
    __shared__ __align__(16) unsigned short lA[128 * GEMM_LDT];
    __shared__ __align__(16) unsigned short lB[128 * GEMM_LDT];
    const int lrow = tid >> 1;        // 0..127
    const int lseg = (tid & 1) * 16;  // 0 or 16 shorts (32B)
    for (int kb = 0; kb < K; kb += 32) {
        const uint4* ga =
            reinterpret_cast<const uint4*>(&A[(size_t)(m0 + lrow) * K + kb + lseg]);
        uint4 a0 = ga[0], a1 = ga[1];
        const uint4* gb =
            reinterpret_cast<const uint4*>(&Bt[(size_t)(n0 + lrow) * K + kb + lseg]);
        uint4 b0 = gb[0], b1 = gb[1];
        __syncthreads();
        *reinterpret_cast<uint4*>(&lA[lrow * GEMM_LDT + lseg])     = a0;
        *reinterpret_cast<uint4*>(&lA[lrow * GEMM_LDT + lseg + 8]) = a1;
        *reinterpret_cast<uint4*>(&lB[lrow * GEMM_LDT + lseg])     = b0;
        *reinterpret_cast<uint4*>(&lB[lrow * GEMM_LDT + lseg + 8]) = b1;
        __syncthreads();
        gemm_step(lA, lB, acc, waveM, waveN, lo, half);
    }
#endif

    // Epilogue: C layout M = r + 8*half, N = lo
#pragma unroll
    for (int mf = 0; mf < 2; ++mf) {
#pragma unroll
        for (int nf = 0; nf < 4; ++nf) {
            int col  = n0 + waveN * 64 + nf * 16 + lo;
            float bv = bias[col];
#pragma unroll
            for (int r = 0; r < 8; ++r) {
                int   row = m0 + waveM * 32 + mf * 16 + r + 8 * half;
                float v   = acc[mf][nf][r] + bv;
                if (OUT_F32)
                    outF[(size_t)row * N + col] = v;
                else
                    outB[(size_t)row * N + col] = f2bf(v);
            }
        }
    }
}

// ---------------------------------------------------------------------------
// Flash attention (causal). qkv bf16 [B*S][3072] (q|k|v), out bf16 [B*S][1024].
// Block: 128 threads (4 waves), 64 q-rows per block, k-tiles of 64.
// Hd = 64 -> 2 WMMA K-steps per contraction; 16 wmma / wave / k-tile.
// Q and K tiles staged by TDM (row 128B + 16B pad -> 72-short stride); the
// V transpose is done manually and overlaps the K-tile DMA.
// ---------------------------------------------------------------------------
__global__ __launch_bounds__(128) void flash_attn_wmma(
    const unsigned short* __restrict__ qkv, unsigned short* __restrict__ attn_out,
    int S) {
    constexpr int LDQ = 72;   // padded stride (144B: 16B aligned, 36-bank stride)
    constexpr float L2E = 1.4426950408889634f;
    __shared__ __align__(16) unsigned short lQ[64 * LDQ];
    __shared__ __align__(16) unsigned short lK[64 * LDQ];
    __shared__ __align__(16) unsigned short lVt[64 * LDQ];    // transposed [d][s_k]
    __shared__ __align__(16) unsigned short lP[4 * 16 * LDQ]; // per-wave P relayout

    const int tid  = threadIdx.x;
    const int lane = tid & 31;
    const int lo   = lane & 15;
    const int half = lane >> 4;
    const int wave = tid >> 5; // 0..3
    const int qt = blockIdx.x;
    const int h  = blockIdx.y;
    const int b  = blockIdx.z;
    const int q0 = qt * 64;

    const size_t rs = 3072;
    const unsigned short* qbase = qkv + (size_t)b * S * rs + h * 64;
    const unsigned short* kbase = qbase + 1024;
    const unsigned short* vbase = qbase + 2048;

    // stage Q tile (64 rows x 64 d). pad codes: interval 4 -> 32 dw, amount 3 -> 4.
#if USE_TDM
    if (tid < 32)
        tdm_load_2d(&qbase[(size_t)q0 * rs], lds_byte_off(lQ),
                    64, 64, 64, 64, (u32)rs, 4, 3);
#else
#pragma unroll
    for (int i = 0; i < 4; ++i) {
        int idx = tid + i * 128;          // 0..511 -> 8-element segments
        int sr  = idx >> 3;
        int d8  = (idx & 7) * 8;
        uint4 v = *reinterpret_cast<const uint4*>(&qbase[(size_t)(q0 + sr) * rs + d8]);
        *reinterpret_cast<uint4*>(&lQ[sr * LDQ + d8]) = v;
    }
#endif

    f32x8 o[4] = {};
    float mrow[8], lrow[8];
#pragma unroll
    for (int r = 0; r < 8; ++r) { mrow[r] = -3.0e38f; lrow[r] = 0.0f; }

    for (int kt = 0; kt <= qt; ++kt) {
        const int k0 = kt * 64;
        __syncthreads(); // everyone done with previous K/V tiles
#if USE_TDM
        if (tid < 32)
            tdm_load_2d(&kbase[(size_t)k0 * rs], lds_byte_off(lK),
                        64, 64, 64, 64, (u32)rs, 4, 3);
#endif
#pragma unroll
        for (int i = 0; i < 4; ++i) {
            int idx = tid + i * 128;
            int sr  = idx >> 3;
            int d8  = (idx & 7) * 8;
#if !USE_TDM
            uint4 kv = *reinterpret_cast<const uint4*>(&kbase[(size_t)(k0 + sr) * rs + d8]);
            *reinterpret_cast<uint4*>(&lK[sr * LDQ + d8]) = kv;
#endif
            uint4 vv = *reinterpret_cast<const uint4*>(&vbase[(size_t)(k0 + sr) * rs + d8]);
            const unsigned short* ve = reinterpret_cast<const unsigned short*>(&vv);
#pragma unroll
            for (int j = 0; j < 8; ++j) lVt[(d8 + j) * LDQ + sr] = ve[j]; // transpose
        }
#if USE_TDM
        if (tid < 32) __builtin_amdgcn_s_wait_tensorcnt(0); // Q (first iter) + K
#endif
        __syncthreads();

        // ---- S = (Q K^T) * 1/sqrt(Hd) -------------------------------------
        f32x8 s[4] = {};
#pragma unroll
        for (int ks = 0; ks < 2; ++ks) {
            Frag aq;
            int qr = wave * 16 + lo;
            aq.h[0] = *reinterpret_cast<const bf16x8*>(&lQ[qr * LDQ + ks * 32 + half * 8]);
            aq.h[1] = *reinterpret_cast<const bf16x8*>(&lQ[qr * LDQ + ks * 32 + 16 + half * 8]);
#pragma unroll
            for (int nf = 0; nf < 4; ++nf) {
                Frag bk;
                int kr = nf * 16 + lo;
                bk.h[0] = *reinterpret_cast<const bf16x8*>(&lK[kr * LDQ + ks * 32 + half * 8]);
                bk.h[1] = *reinterpret_cast<const bf16x8*>(&lK[kr * LDQ + ks * 32 + 16 + half * 8]);
                s[nf] = wmma_bf16(aq.v, bk.v, s[nf]);
            }
        }
        const bool diag = (kt == qt);
#pragma unroll
        for (int nf = 0; nf < 4; ++nf) {
#pragma unroll
            for (int r = 0; r < 8; ++r) {
                float v = s[nf][r] * 0.125f; // 1/sqrt(64), exact
                if (diag) {
                    int qrow = q0 + wave * 16 + r + 8 * half;
                    int kcol = k0 + nf * 16 + lo;
                    if (kcol > qrow) v = -3.0e38f;
                }
                s[nf][r] = v;
            }
        }

        // ---- online softmax update ----------------------------------------
        float mnew[8], alpha[8], psum[8];
#pragma unroll
        for (int r = 0; r < 8; ++r) {
            float mx = fmaxf(fmaxf(s[0][r], s[1][r]), fmaxf(s[2][r], s[3][r]));
#pragma unroll
            for (int off = 1; off < 16; off <<= 1)
                mx = fmaxf(mx, __shfl_xor(mx, off, 32)); // stays in 16-lane half
            mnew[r]  = fmaxf(mrow[r], mx);
            alpha[r] = __builtin_exp2f((mrow[r] - mnew[r]) * L2E);
            mrow[r]  = mnew[r];
            psum[r]  = 0.0f;
        }
#pragma unroll
        for (int nf = 0; nf < 4; ++nf) {
#pragma unroll
            for (int r = 0; r < 8; ++r) {
                float p = __builtin_exp2f((s[nf][r] - mnew[r]) * L2E);
                psum[r] += p;
                // C-layout -> A-layout relayout through per-wave LDS region
                lP[(wave * 16 + r + 8 * half) * LDQ + nf * 16 + lo] = f2bf(p);
            }
        }
#pragma unroll
        for (int r = 0; r < 8; ++r) {
            float ps = psum[r];
#pragma unroll
            for (int off = 1; off < 16; off <<= 1) ps += __shfl_xor(ps, off, 32);
            lrow[r] = lrow[r] * alpha[r] + ps;
        }
#pragma unroll
        for (int f = 0; f < 4; ++f)
#pragma unroll
            for (int r = 0; r < 8; ++r) o[f][r] *= alpha[r];

        // ---- O += P V  (LDS ops are in-order within a wave: no barrier) ----
#pragma unroll
        for (int ks = 0; ks < 2; ++ks) {
            Frag ap;
            int pr = wave * 16 + lo;
            ap.h[0] = *reinterpret_cast<const bf16x8*>(&lP[pr * LDQ + ks * 32 + half * 8]);
            ap.h[1] = *reinterpret_cast<const bf16x8*>(&lP[pr * LDQ + ks * 32 + 16 + half * 8]);
#pragma unroll
            for (int df = 0; df < 4; ++df) {
                Frag bv;
                int dr = df * 16 + lo;
                bv.h[0] = *reinterpret_cast<const bf16x8*>(&lVt[dr * LDQ + ks * 32 + half * 8]);
                bv.h[1] = *reinterpret_cast<const bf16x8*>(&lVt[dr * LDQ + ks * 32 + 16 + half * 8]);
                o[df] = wmma_bf16(ap.v, bv.v, o[df]);
            }
        }
    }

    // ---- normalize + merge-heads store (bf16) -----------------------------
#pragma unroll
    for (int df = 0; df < 4; ++df) {
        int col = h * 64 + df * 16 + lo;
#pragma unroll
        for (int r = 0; r < 8; ++r) {
            int row = b * S + q0 + wave * 16 + r + 8 * half;
            attn_out[(size_t)row * 1024 + col] = f2bf(o[df][r] / lrow[r]);
        }
    }
}

// ---------------------------------------------------------------------------
// Launch
// ---------------------------------------------------------------------------
extern "C" void kernel_launch(void* const* d_in, const int* in_sizes, int n_in,
                              void* d_out, int out_size, void* d_ws, size_t ws_size,
                              hipStream_t stream) {
    (void)in_sizes; (void)n_in; (void)out_size; (void)ws_size;
    const float* X  = (const float*)d_in[0]; // [2,2048,1024]
    const float* Wa = (const float*)d_in[1]; // [1024,3072]
    const float* ba = (const float*)d_in[2]; // [3072]
    const float* Wp = (const float*)d_in[3]; // [1024,1024]
    const float* bp = (const float*)d_in[4]; // [1024]
    float* out = (float*)d_out;              // [2,2048,1024]

    constexpr int BS = 2 * 2048; // 4096 rows
    // Workspace layout (bf16 shorts), total 48 MB:
    unsigned short* xb  = (unsigned short*)d_ws;       // X bf16      [4096*1024]
    unsigned short* wat = xb  + (size_t)BS * 1024;     // Wattn^T     [3072*1024]
    unsigned short* wpt = wat + (size_t)3072 * 1024;   // Wproj^T     [1024*1024]
    unsigned short* qkv = wpt + (size_t)1024 * 1024;   // QKV bf16    [4096*3072]
    unsigned short* ao  = qkv + (size_t)BS * 3072;     // attn out    [4096*1024]

    cvt_f32_to_bf16<<<(BS * 1024) / 256, 256, 0, stream>>>(X, xb, BS * 1024);
    cvt_transpose_1024<<<(3072 * 1024) / 256, 256, 0, stream>>>(Wa, wat, 3072);
    cvt_transpose_1024<<<(1024 * 1024) / 256, 256, 0, stream>>>(Wp, wpt, 1024);

    // QKV = X @ Wa + ba  (bf16 out)
    gemm_bf16_wmma<false><<<dim3(3072 / 128, BS / 128), 256, 0, stream>>>(
        xb, wat, ba, nullptr, qkv, BS, 3072, 1024);

    // flash attention, causal
    flash_attn_wmma<<<dim3(2048 / 64, 16, 2), 128, 0, stream>>>(qkv, ao, 2048);

    // out = attn @ Wp + bp  (f32 out)
    gemm_bf16_wmma<true><<<dim3(1024 / 128, BS / 128), 256, 0, stream>>>(
        ao, wpt, bp, out, nullptr, BS, 1024, 1024);
}
